// DeepCoevolve_45432164057526
// MI455X (gfx1250) — matrix operand order repaired and checked by reading, not compiled, not run on hardware
//
#include <hip/hip_runtime.h>
#include <math.h>

// ---------------------------------------------------------------------------
// Types for CDNA5 WMMA (wave32)
// ---------------------------------------------------------------------------
typedef __attribute__((ext_vector_type(16))) __bf16 v16bf;
typedef __attribute__((ext_vector_type(8)))  __bf16 v8bf;
typedef __attribute__((ext_vector_type(8)))  float  v8f;

#define E_EVENTS 4096
#define P_PATHS  8
#define HDIM     128
#define NSEQ     (E_EVENTS * P_PATHS)   // 32768

__device__ __forceinline__ float sigm(float x) { return 1.0f / (1.0f + expf(-x)); }

// A-matrix fragment (16x32 bf16, MxK): lane m = lane%16 row;
// K chunks: [8h, 8h+8) and [8h+16, 8h+24), h = lane>>4.
// rowbase must point at element [row][0]; kk is the K-tile base.
__device__ __forceinline__ v16bf load_a_frag(const __bf16* rowbase, int kk, int lane) {
    int khalf = (lane >> 4) * 8;
    const __bf16* p = rowbase + kk + khalf;
    v16bf a;
    *(v8bf*)&a       = *(const v8bf*)p;
    *((v8bf*)&a + 1) = *(const v8bf*)(p + 16);
    return a;
}

// B-matrix fragment (32x16 bf16, KxN) loaded from row-major W (N x K), i.e. B^T.
// lane n = lane%16 column; K chunks: [8h, 8h+8) and [8h+16, 8h+24), h = lane>>4.
__device__ __forceinline__ v16bf load_b_frag(const __bf16* wrowbase, int kk, int lane) {
    int koff = kk + ((lane >> 4) << 3);
    const __bf16* p = wrowbase + koff;
    v16bf b;
    *(v8bf*)&b       = *(const v8bf*)p;
    *((v8bf*)&b + 1) = *(const v8bf*)(p + 16);
    return b;
}

__device__ __forceinline__ v8f wmma_bf16(v16bf a, v16bf b, v8f c) {
    return __builtin_amdgcn_wmma_f32_16x16x32_bf16(false, a, false, b, (short)0, c, false, false);
}

// ---------------------------------------------------------------------------
// Weight conversion kernels
// ---------------------------------------------------------------------------
__global__ void k_cvt_bf16(const float* __restrict__ in, __bf16* __restrict__ out, int n) {
    int i = blockIdx.x * blockDim.x + threadIdx.x;
    if (i < n) out[i] = (__bf16)in[i];
}

// Build [Wih | Whh] (rows x 256) bf16, each input rows x 128
__global__ void k_concat_w(const float* __restrict__ A, const float* __restrict__ B,
                           __bf16* __restrict__ out, int rows) {
    int i = blockIdx.x * blockDim.x + threadIdx.x;
    if (i >= rows * 256) return;
    int r = i >> 8, c = i & 255;
    float v = (c < 128) ? A[r * 128 + c] : B[r * 128 + (c - 128)];
    out[i] = (__bf16)v;
}

// ---------------------------------------------------------------------------
// Gather kernels
// ---------------------------------------------------------------------------
__global__ __launch_bounds__(128) void k_gather_seq(const int* __restrict__ path_idx,
                                                    const float* __restrict__ uemb,
                                                    const float* __restrict__ iemb,
                                                    __bf16* __restrict__ xseq) {
    int n = blockIdx.x;          // path id = e*8+p
    int c = threadIdx.x;         // 0..127
    const int* idx4 = path_idx + (size_t)n * 4;
    xseq[((size_t)0 * NSEQ + n) * HDIM + c] = (__bf16)uemb[(size_t)idx4[0] * HDIM + c];
    xseq[((size_t)1 * NSEQ + n) * HDIM + c] = (__bf16)iemb[(size_t)idx4[1] * HDIM + c];
    xseq[((size_t)2 * NSEQ + n) * HDIM + c] = (__bf16)uemb[(size_t)idx4[2] * HDIM + c];
    xseq[((size_t)3 * NSEQ + n) * HDIM + c] = (__bf16)iemb[(size_t)idx4[3] * HDIM + c];
}

__global__ void k_gather_events(const int* __restrict__ eu, const int* __restrict__ ei,
                                const float* __restrict__ uemb, const float* __restrict__ iemb,
                                float* __restrict__ ue_f, float* __restrict__ ie_f,
                                __bf16* __restrict__ ue_b, __bf16* __restrict__ ie_b) {
    int idx = blockIdx.x * blockDim.x + threadIdx.x;
    if (idx >= E_EVENTS * HDIM) return;
    int e = idx >> 7, c = idx & 127;
    float u = uemb[(size_t)eu[e] * HDIM + c];
    float v = iemb[(size_t)ei[e] * HDIM + c];
    ue_f[idx] = u; ie_f[idx] = v;
    ue_b[idx] = (__bf16)u; ie_b[idx] = (__bf16)v;
}

// ---------------------------------------------------------------------------
// Fused LSTM step: g = [x_t, h] @ [Wih|Whh]^T + bih + bhh, then gate update.
// Block: 16 rows x full 512 cols, 8 waves (each wave: 4 col-tiles, K=256).
// ---------------------------------------------------------------------------
__global__ __launch_bounds__(256) void k_lstm_step(const __bf16* __restrict__ x_t,  // NSEQ x 128
                                                   __bf16* __restrict__ h_bf,       // NSEQ x 128 in/out
                                                   float* __restrict__ c_f,         // NSEQ x 128 in/out
                                                   float* __restrict__ h_f,         // NSEQ x 128 out
                                                   const __bf16* __restrict__ Wc,   // 512 x 256
                                                   const float* __restrict__ bih,
                                                   const float* __restrict__ bhh) {
    __shared__ float gbuf[16][512];
    const int tid  = threadIdx.x;
    const int wave = tid >> 5;
    const int lane = tid & 31;
    const int r0   = blockIdx.x * 16;
    const int mrow = lane & 15;

    v8f acc[4] = {};
    const __bf16* xrow = x_t  + (size_t)(r0 + mrow) * HDIM;
    const __bf16* hrow = h_bf + (size_t)(r0 + mrow) * HDIM;

#pragma unroll
    for (int kk = 0; kk < 256; kk += 32) {
        v16bf a = (kk < 128) ? load_a_frag(xrow, kk, lane)
                             : load_a_frag(hrow, kk - 128, lane);
#pragma unroll
        for (int j = 0; j < 4; ++j) {
            int n = wave * 64 + j * 16 + mrow;
            v16bf b = load_b_frag(Wc + (size_t)n * 256, kk, lane);
            acc[j] = wmma_bf16(a, b, acc[j]);
        }
    }

#pragma unroll
    for (int j = 0; j < 4; ++j) {
        int col = wave * 64 + j * 16 + mrow;
        float bias = bih[col] + bhh[col];
#pragma unroll
        for (int v = 0; v < 8; ++v) {
            int m = v + ((lane >> 4) << 3);
            gbuf[m][col] = acc[j][v] + bias;
        }
    }
    __syncthreads();

#pragma unroll
    for (int i = 0; i < 8; ++i) {
        int idx = tid + i * 256;          // 0..2047 over 16x128
        int m = idx >> 7, hc = idx & 127;
        size_t gofs = (size_t)(r0 + m) * HDIM + hc;
        float ig = gbuf[m][hc];
        float fg = gbuf[m][hc + 128];
        float gg = gbuf[m][hc + 256];
        float og = gbuf[m][hc + 384];
        float cn = sigm(fg) * c_f[gofs] + sigm(ig) * tanhf(gg);
        float hn = sigm(og) * tanhf(cn);
        c_f[gofs] = cn;
        h_f[gofs] = hn;
        h_bf[gofs] = (__bf16)hn;
    }
}

// ---------------------------------------------------------------------------
// Generic bf16 WMMA GEMM: C(MxN,f32) = A(MxK) @ W(NxK)^T + bias(N)
// Block tile 64x32; 8 waves arranged 4(M) x 2(N), one 16x16 tile each.
// ---------------------------------------------------------------------------
__global__ __launch_bounds__(256) void k_gemm_bf16(const __bf16* __restrict__ A,
                                                   const __bf16* __restrict__ W,
                                                   const float* __restrict__ bias,
                                                   float* __restrict__ C,
                                                   int N, int K) {
    const int tid  = threadIdx.x;
    const int wave = tid >> 5;
    const int lane = tid & 31;
    const int wm = wave >> 1, wn = wave & 1;
    const int m0 = blockIdx.x * 64 + wm * 16;
    const int n0 = blockIdx.y * 32 + wn * 16;
    const int mrow = lane & 15;

    v8f acc = {};
    const __bf16* arow = A + (size_t)(m0 + mrow) * K;
    const __bf16* wrow = W + (size_t)(n0 + mrow) * K;
    for (int kk = 0; kk < K; kk += 32) {
        v16bf a = load_a_frag(arow, kk, lane);
        v16bf b = load_b_frag(wrow, kk, lane);
        acc = wmma_bf16(a, b, acc);
    }
    int n = n0 + mrow;
    float bv = bias[n];
#pragma unroll
    for (int v = 0; v < 8; ++v) {
        int m = m0 + v + ((lane >> 4) << 3);
        C[(size_t)m * N + n] = acc[v] + bv;
    }
}

// ---------------------------------------------------------------------------
// Attention over P=8 paths: att = softmax_p(max_h h[e,p,:]); path_emb -> bf16
// written into cols [128,256) of both concat buffers. One block per event.
// ---------------------------------------------------------------------------
__global__ __launch_bounds__(256) void k_attention(const float* __restrict__ h_f,
                                                   __bf16* __restrict__ ucat,
                                                   __bf16* __restrict__ icat) {
    int e = blockIdx.x;
    int tid = threadIdx.x, wave = tid >> 5, lane = tid & 31;
    __shared__ float mx[8];
    __shared__ float att[8];

    const float* hp = h_f + ((size_t)e * 8 + wave) * HDIM;
    float m = -1e30f;
    for (int i = lane; i < HDIM; i += 32) m = fmaxf(m, hp[i]);
#pragma unroll
    for (int off = 16; off > 0; off >>= 1) m = fmaxf(m, __shfl_xor(m, off, 32));
    if (lane == 0) mx[wave] = m;
    __syncthreads();
    if (tid == 0) {
        float mm = -1e30f;
        for (int p = 0; p < 8; ++p) mm = fmaxf(mm, mx[p]);
        float s = 0.f, ex[8];
        for (int p = 0; p < 8; ++p) { ex[p] = expf(mx[p] - mm); s += ex[p]; }
        for (int p = 0; p < 8; ++p) att[p] = ex[p] / s;
    }
    __syncthreads();
    if (tid < HDIM) {
        float acc = 0.f;
        for (int p = 0; p < 8; ++p)
            acc += att[p] * h_f[((size_t)e * 8 + p) * HDIM + tid];
        __bf16 b = (__bf16)acc;
        ucat[(size_t)e * 256 + 128 + tid] = b;
        icat[(size_t)e * 256 + 128 + tid] = b;
    }
}

// ---------------------------------------------------------------------------
// GRU pointwise: r=sig(ir+hr), z=sig(iz+hz), n=tanh(inn+r*hn); out=(1-z)n+z*h
// Writes bf16 result into cols [0,128) of concat buffer.
// ---------------------------------------------------------------------------
__global__ void k_gru_pw(const float* __restrict__ gi, const float* __restrict__ gh,
                         const float* __restrict__ h, __bf16* __restrict__ cat) {
    int idx = blockIdx.x * blockDim.x + threadIdx.x;
    if (idx >= E_EVENTS * HDIM) return;
    int e = idx >> 7, hc = idx & 127;
    size_t base = (size_t)e * 384 + hc;
    float ir = gi[base], iz = gi[base + 128], inn = gi[base + 256];
    float hr = gh[base], hz = gh[base + 128], hn  = gh[base + 256];
    float r = sigm(ir + hr);
    float z = sigm(iz + hz);
    float nt = tanhf(inn + r * hn);
    float out = (1.0f - z) * nt + z * h[idx];
    cat[(size_t)e * 256 + hc] = (__bf16)out;
}

// ---------------------------------------------------------------------------
// Final: comp[e] = softplus(dot(u_fused[e], i_fused[e]))
// ---------------------------------------------------------------------------
__global__ __launch_bounds__(128) void k_final(const float* __restrict__ uf,
                                               const float* __restrict__ vf,
                                               float* __restrict__ out) {
    int e = blockIdx.x, tid = threadIdx.x;
    float p = uf[(size_t)e * HDIM + tid] * vf[(size_t)e * HDIM + tid];
#pragma unroll
    for (int off = 16; off > 0; off >>= 1) p += __shfl_xor(p, off, 32);
    __shared__ float s[4];
    if ((tid & 31) == 0) s[tid >> 5] = p;
    __syncthreads();
    if (tid == 0) {
        float x = s[0] + s[1] + s[2] + s[3];
        out[e] = (x > 20.0f) ? x : log1pf(expf(x));
    }
}

// ---------------------------------------------------------------------------
// Launch
// ---------------------------------------------------------------------------
extern "C" void kernel_launch(void* const* d_in, const int* in_sizes, int n_in,
                              void* d_out, int out_size, void* d_ws, size_t ws_size,
                              hipStream_t stream) {
    (void)in_sizes; (void)n_in; (void)out_size; (void)ws_size;

    const int*   path_idx   = (const int*)d_in[0];
    const int*   event_user = (const int*)d_in[1];
    const int*   event_item = (const int*)d_in[2];
    const float* user_emb   = (const float*)d_in[3];
    const float* item_emb   = (const float*)d_in[4];
    const float* lstm_Wih   = (const float*)d_in[5];
    const float* lstm_Whh   = (const float*)d_in[6];
    const float* lstm_bih   = (const float*)d_in[7];
    const float* lstm_bhh   = (const float*)d_in[8];
    const float* gu_Wih = (const float*)d_in[9];
    const float* gu_Whh = (const float*)d_in[10];
    const float* gu_bih = (const float*)d_in[11];
    const float* gu_bhh = (const float*)d_in[12];
    const float* gI_Wih = (const float*)d_in[13];
    const float* gI_Whh = (const float*)d_in[14];
    const float* gI_bih = (const float*)d_in[15];
    const float* gI_bhh = (const float*)d_in[16];
    const float* ulW = (const float*)d_in[17];
    const float* ulb = (const float*)d_in[18];
    const float* ilW = (const float*)d_in[19];
    const float* ilb = (const float*)d_in[20];
    float* out = (float*)d_out;

    char* w = (char*)d_ws;
    size_t off = 0;
    auto take = [&](size_t b) -> char* {
        char* p = w + off;
        off += (b + 255) & ~(size_t)255;
        return p;
    };

    __bf16* lstmW  = (__bf16*)take(512 * 256 * 2);
    __bf16* guWihB = (__bf16*)take(384 * 128 * 2);
    __bf16* guWhhB = (__bf16*)take(384 * 128 * 2);
    __bf16* gIWihB = (__bf16*)take(384 * 128 * 2);
    __bf16* gIWhhB = (__bf16*)take(384 * 128 * 2);
    __bf16* ulWB   = (__bf16*)take(128 * 256 * 2);
    __bf16* ilWB   = (__bf16*)take(128 * 256 * 2);
    __bf16* xseq   = (__bf16*)take((size_t)4 * NSEQ * HDIM * 2);   // 33.5 MB
    __bf16* h_bf   = (__bf16*)take((size_t)NSEQ * HDIM * 2);
    float*  h_f    = (float*)take((size_t)NSEQ * HDIM * 4);
    float*  c_f    = (float*)take((size_t)NSEQ * HDIM * 4);
    float*  ue_f   = (float*)take((size_t)E_EVENTS * HDIM * 4);
    float*  ie_f   = (float*)take((size_t)E_EVENTS * HDIM * 4);
    __bf16* ue_b   = (__bf16*)take((size_t)E_EVENTS * HDIM * 2);
    __bf16* ie_b   = (__bf16*)take((size_t)E_EVENTS * HDIM * 2);
    __bf16* ucat   = (__bf16*)take((size_t)E_EVENTS * 256 * 2);
    __bf16* icat   = (__bf16*)take((size_t)E_EVENTS * 256 * 2);
    float*  ufused = (float*)take((size_t)E_EVENTS * HDIM * 4);
    float*  ifused = (float*)take((size_t)E_EVENTS * HDIM * 4);

    // GRU gate scratch overlays xseq (dead after the last LSTM step)
    float* giu = (float*)((char*)xseq + 0);
    float* ghu = (float*)((char*)xseq + (size_t)1 * E_EVENTS * 384 * 4);
    float* gii = (float*)((char*)xseq + (size_t)2 * E_EVENTS * 384 * 4);
    float* ghi = (float*)((char*)xseq + (size_t)3 * E_EVENTS * 384 * 4);

    // --- init state ---
    hipMemsetAsync(h_bf, 0, (size_t)NSEQ * HDIM * 2, stream);
    hipMemsetAsync(c_f,  0, (size_t)NSEQ * HDIM * 4, stream);

    // --- weight conversion ---
    k_concat_w<<<512, 256, 0, stream>>>(lstm_Wih, lstm_Whh, lstmW, 512);
    k_cvt_bf16<<<(384 * 128 + 255) / 256, 256, 0, stream>>>(gu_Wih, guWihB, 384 * 128);
    k_cvt_bf16<<<(384 * 128 + 255) / 256, 256, 0, stream>>>(gu_Whh, guWhhB, 384 * 128);
    k_cvt_bf16<<<(384 * 128 + 255) / 256, 256, 0, stream>>>(gI_Wih, gIWihB, 384 * 128);
    k_cvt_bf16<<<(384 * 128 + 255) / 256, 256, 0, stream>>>(gI_Whh, gIWhhB, 384 * 128);
    k_cvt_bf16<<<(128 * 256 + 255) / 256, 256, 0, stream>>>(ulW, ulWB, 128 * 256);
    k_cvt_bf16<<<(128 * 256 + 255) / 256, 256, 0, stream>>>(ilW, ilWB, 128 * 256);

    // --- gathers ---
    k_gather_seq<<<NSEQ, 128, 0, stream>>>(path_idx, user_emb, item_emb, xseq);
    k_gather_events<<<(E_EVENTS * HDIM + 255) / 256, 256, 0, stream>>>(
        event_user, event_item, user_emb, item_emb, ue_f, ie_f, ue_b, ie_b);

    // --- LSTM: 4 sequential fused steps ---
    for (int t = 0; t < 4; ++t) {
        k_lstm_step<<<NSEQ / 16, 256, 0, stream>>>(
            xseq + (size_t)t * NSEQ * HDIM, h_bf, c_f, h_f, lstmW, lstm_bih, lstm_bhh);
    }

    // --- attention / path pooling (writes cat cols 128..255) ---
    k_attention<<<E_EVENTS, 256, 0, stream>>>(h_f, ucat, icat);

    // --- GRU gate GEMMs (M=4096, N=384, K=128) ---
    dim3 gG(E_EVENTS / 64, 384 / 32);
    k_gemm_bf16<<<gG, 256, 0, stream>>>(ie_b, guWihB, gu_bih, giu, 384, 128); // gi_u: x=i_e
    k_gemm_bf16<<<gG, 256, 0, stream>>>(ue_b, guWhhB, gu_bhh, ghu, 384, 128); // gh_u: h=u_e
    k_gemm_bf16<<<gG, 256, 0, stream>>>(ue_b, gIWihB, gI_bih, gii, 384, 128); // gi_i: x=u_e
    k_gemm_bf16<<<gG, 256, 0, stream>>>(ie_b, gIWhhB, gI_bhh, ghi, 384, 128); // gh_i: h=i_e

    // --- GRU pointwise (writes cat cols 0..127) ---
    k_gru_pw<<<(E_EVENTS * HDIM + 255) / 256, 256, 0, stream>>>(giu, ghu, ue_f, ucat);
    k_gru_pw<<<(E_EVENTS * HDIM + 255) / 256, 256, 0, stream>>>(gii, ghi, ie_f, icat);

    // --- fusion linears (M=4096, N=128, K=256) ---
    dim3 gF(E_EVENTS / 64, 128 / 32);
    k_gemm_bf16<<<gF, 256, 0, stream>>>(ucat, ulWB, ulb, ufused, 128, 256);
    k_gemm_bf16<<<gF, 256, 0, stream>>>(icat, ilWB, ilb, ifused, 128, 256);

    // --- final dot + softplus ---
    k_final<<<E_EVENTS, 128, 0, stream>>>(ufused, ifused, out);
}
